// FullColumn_17214228922888
// MI455X (gfx1250) — compile-verified, matrix operand order checked by reading, not run on hardware
//
#include <hip/hip_runtime.h>
#include <hip/hip_bf16.h>

typedef __bf16 bf16_t;
typedef __attribute__((ext_vector_type(16))) __bf16 v16bf;
typedef __attribute__((ext_vector_type(8)))  float  v8f;
typedef __attribute__((ext_vector_type(4)))  int    v4i;

#define NBATCH   16
#define NCH      512          // synapses (contraction channels)
#define T_IN     512          // input timesteps
#define NEUR     1024         // output neurons (M)
#define KS       48           // temporal kernel taps
#define T_OUT    529          // 512 + 2*32 - 48 + 1
#define PADT     32
#define T_PAD    640          // 32 front pad + 512 + 96 back pad (time-major x)
#define THETA    25.6f
#define BIAS_POT 12.8f        // BIAS * THETA

// workspace layout (bytes)
#define WK2_BYTES  (1024UL * 48 * 512 * 2)          // 50,331,648  bf16 kernel matrix
#define XBT_BYTES  (16UL * T_PAD * 512 * 2)         // 10,485,760  bf16 spikes, [b][t_pad][ch]
#define POT_OFF    (WK2_BYTES + XBT_BYTES)
// pot: 16*1024*529 f32 = 34,668,544 bytes ; total ws use ~95.5 MB

// ---- CDNA5 async global->LDS path (ASYNCcnt), with sync fallback -----------
#if defined(__AMDGCN__) && __has_builtin(__builtin_amdgcn_global_load_async_to_lds_b128)
#define USE_ASYNC 1
typedef v4i __attribute__((address_space(1))) as1_v4i;   // global-AS 16B vector
typedef v4i __attribute__((address_space(3))) as3_v4i;   // LDS-AS 16B vector
#endif

__device__ __forceinline__ void wait_async_lds() {
#ifdef USE_ASYNC
#if __has_builtin(__builtin_amdgcn_s_wait_asynccnt)
    __builtin_amdgcn_s_wait_asynccnt(0);
#else
    asm volatile("s_wait_asynccnt 0x0" ::: "memory");
#endif
#endif
}

// ---------------------------------------------------------------------------
// Phase 1a: build bf16 kernel matrix Wk2[o][kk][i] (conv-flipped tap order)
// kernel_raw(k) = max(0, min(k/16, -(k - w*16)/32 + w)), stored at kk = 47-k
// ---------------------------------------------------------------------------
__global__ void build_wk(const float* __restrict__ weight, bf16_t* __restrict__ wk2) {
    size_t total = 1024UL * 48 * 512;
    for (size_t idx = (size_t)blockIdx.x * blockDim.x + threadIdx.x; idx < total;
         idx += (size_t)gridDim.x * blockDim.x) {
        int i  = (int)(idx % 512);
        int kk = (int)((idx / 512) % 48);
        int o  = (int)(idx / (512 * 48));
        float w  = weight[o * 512 + i];
        int   k  = 47 - kk;                                   // flip for correlation
        float ts = (float)k * (1.0f / 16.0f);
        float tl = -((float)k - w * 16.0f) * (1.0f / 32.0f) + w;
        float v  = fmaxf(0.0f, fminf(ts, tl));
        wk2[idx] = (bf16_t)v;
    }
}

// Phase 1b: spikes f32 (b,ch,t) -> bf16 time-major padded (b, t_pad, ch).
// t_pad in [0,640): zero for t_pad<32 or >=544, else spikes[b,ch,t_pad-32].
__global__ void cvt_x(const float* __restrict__ xs, bf16_t* __restrict__ xbt) {
    size_t total = (size_t)NBATCH * T_PAD * NCH;
    for (size_t idx = (size_t)blockIdx.x * blockDim.x + threadIdx.x; idx < total;
         idx += (size_t)gridDim.x * blockDim.x) {
        int ch = (int)(idx % NCH);
        int tp = (int)((idx / NCH) % T_PAD);
        int b  = (int)(idx / ((size_t)NCH * T_PAD));
        float v = 0.0f;
        int t = tp - PADT;
        if (t >= 0 && t < T_IN) v = xs[((size_t)b * NCH + ch) * T_IN + t];
        xbt[idx] = (bf16_t)v;
    }
}

// ---------------------------------------------------------------------------
// Tile fill: copy xbt[b][t0 .. t0+112)[ib .. ib+32) into LDS buffer [112][32].
// 448 contiguous 16-byte chunks; async DMA on CDNA5, vectorized copy fallback.
// ---------------------------------------------------------------------------
__device__ __forceinline__ void fill_tile(bf16_t (*sx)[32], const bf16_t* __restrict__ xbt,
                                          int b, int t0, int ib, int tid) {
    #pragma unroll
    for (int r = 0; r < 4; ++r) {
        int idx = tid + r * 128;
        if (idx < 112 * 4) {
            int tt = idx >> 2;
            int c8 = (idx & 3) << 3;
            const bf16_t* g = xbt + ((size_t)b * T_PAD + t0 + tt) * NCH + ib + c8;
            bf16_t*       l = &sx[tt][c8];
#ifdef USE_ASYNC
            // flat global address == AS(1) address; low 32 bits of a flat LDS
            // address are the LDS byte offset (aperture truncation rule).
            as1_v4i* gp = (as1_v4i*)(uintptr_t)g;
            as3_v4i* lp = (as3_v4i*)(unsigned)(uintptr_t)l;
            __builtin_amdgcn_global_load_async_to_lds_b128(gp, lp, 0, 0);
#else
            *(uint4*)l = *(const uint4*)g;
#endif
        }
    }
}

// ---------------------------------------------------------------------------
// Phase 2: 1D conv as bf16 WMMA GEMM.
//   pot[b, o, t'] = sum_{kk, i} Wk2[o, kk, i] * x[b, i, t' + kk - 32]
// Block = 128 threads (4 waves). Tile: 64 neurons (16/wave) x 64 timesteps.
// Contraction chunk (K=32) = 32 channels at a fixed tap kk. Double-buffered
// LDS spike window: async-fill next channel block while WMMAs run on current.
// ---------------------------------------------------------------------------
__global__ __launch_bounds__(128)
void snn_conv_wmma(const bf16_t* __restrict__ wk2, const bf16_t* __restrict__ xbt,
                   float* __restrict__ pot) {
    __shared__ __align__(16) bf16_t sbuf[2][112][32];   // 14 KB

    const int tid    = threadIdx.x;
    const int wave   = tid >> 5;          // wave32
    const int lane   = tid & 31;
    const int laneHi = lane >> 4;         // 0/1 half-wave
    const int lml    = lane & 15;
    const int b      = blockIdx.z;
    const int t0     = blockIdx.y * 64;   // padded-time base (= real t' base)
    const int m0     = blockIdx.x * 64 + wave * 16;
    const int orow   = m0 + lml;          // A-matrix row owned by this lane

    v8f acc[4] = {};                      // 16x16 f32 accum per 16-time subtile

    fill_tile(sbuf[0], xbt, b, t0, 0, tid);     // prologue DMA

    int pb = 0;
    for (int ib = 0; ib < NCH; ib += 32, pb ^= 1) {
        wait_async_lds();                 // my async fills for sbuf[pb] done
        __syncthreads();                  // everyone's fills visible; prev reads done
        if (ib + 32 < NCH)
            fill_tile(sbuf[pb ^ 1], xbt, b, t0, ib + 32, tid);   // overlap w/ compute

        bf16_t (*sx)[32] = sbuf[pb];

        // A fragments: Wk2 row (orow), tap kk, channels [ib, ib+32)
        const bf16_t* arow = wk2 + (size_t)orow * KS * NCH + ib + laneHi * 8;

        for (int kk = 0; kk < KS; ++kk) {
            union { v16bf v; uint4 u[2]; } af;
            af.u[0] = *(const uint4*)(arow);        // K = +0..7  (this half-wave)
            af.u[1] = *(const uint4*)(arow + 16);   // K = +16..23
            arow += NCH;                            // next tap

            #pragma unroll
            for (int j = 0; j < 4; ++j) {
                int ttl = j * 16 + lml + kk;        // time row in LDS (<= 110)
                const uint4* srow = (const uint4*)&sx[ttl][0];
                union { v16bf v; uint4 u[2]; } bfr;
                bfr.u[0] = srow[laneHi];            // channels 0..7 / 8..15
                bfr.u[1] = srow[laneHi + 2];        // channels 16..23 / 24..31
                acc[j] = __builtin_amdgcn_wmma_f32_16x16x32_bf16(
                    /*neg_a=*/false, af.v, /*neg_b=*/false, bfr.v,
                    /*c_mod=*/(short)0, acc[j], /*reuse_a=*/false, /*reuse_b=*/false);
            }
        }
    }

    // ---- store pot (+ bias). C layout: VGPR r -> M = r + 8*laneHi, N = lane&15
    #pragma unroll
    for (int j = 0; j < 4; ++j) {
        int t = t0 + j * 16 + lml;
        if (t < T_OUT) {
            int obase  = m0 + laneHi * 8;
            float* dst = pot + ((size_t)b * NEUR + obase) * T_OUT + t;
            #pragma unroll
            for (int r = 0; r < 8; ++r)
                dst[(size_t)r * T_OUT] = acc[j][r] + BIAS_POT;
        }
    }
}

// ---------------------------------------------------------------------------
// Phase 3: sequential winner-takes-all scan. One block per batch, one thread
// per neuron. C=1 so depression reduces to: dep = clip(dep + 48*spike - 1, 0, 47).
// Wave32 shfl-xor butterfly argmax + one cross-wave stage: 2 barriers/step.
// ---------------------------------------------------------------------------
__global__ __launch_bounds__(1024)
void wta_kernel(const float* __restrict__ pot, float* __restrict__ out) {
    const int b    = blockIdx.x;
    const int n    = threadIdx.x;
    const int lane = n & 31;
    const int wv   = n >> 5;
    __shared__ float pval[32];
    __shared__ int   pidx[32];
    __shared__ float sspike;
    __shared__ int   swidx;

    float dep = 0.0f;
    const float* prow = pot + ((size_t)b * NEUR + n) * T_OUT;
    float*       orow = out + ((size_t)b * NEUR + n) * T_OUT;

    for (int t = 0; t < T_OUT; ++t) {
        float v  = (dep == 0.0f) ? prow[t] : 0.0f;   // mask depressed neurons
        int  idx = n;
        #pragma unroll
        for (int m = 16; m > 0; m >>= 1) {           // wave32 argmax (first-max)
            float ov = __shfl_xor(v, m, 32);
            int   oi = __shfl_xor(idx, m, 32);
            if (ov > v || (ov == v && oi < idx)) { v = ov; idx = oi; }
        }
        if (lane == 0) { pval[wv] = v; pidx[wv] = idx; }
        __syncthreads();
        if (wv == 0) {
            float v2 = pval[lane];
            int   i2 = pidx[lane];
            #pragma unroll
            for (int m = 16; m > 0; m >>= 1) {
                float ov = __shfl_xor(v2, m, 32);
                int   oi = __shfl_xor(i2, m, 32);
                if (ov > v2 || (ov == v2 && oi < i2)) { v2 = ov; i2 = oi; }
            }
            if (lane == 0) { sspike = (v2 > THETA) ? 1.0f : 0.0f; swidx = i2; }
        }
        __syncthreads();
        float spike = sspike;
        orow[t] = (spike != 0.0f && n == swidx) ? 1.0f : 0.0f;
        dep = fminf(fmaxf(dep + spike * 48.0f - 1.0f, 0.0f), 47.0f);
    }
}

// ---------------------------------------------------------------------------
extern "C" void kernel_launch(void* const* d_in, const int* in_sizes, int n_in,
                              void* d_out, int out_size, void* d_ws, size_t ws_size,
                              hipStream_t stream) {
    (void)in_sizes; (void)n_in; (void)out_size; (void)ws_size;
    const float* spikes = (const float*)d_in[0];   // (16,1,512,512) f32
    const float* weight = (const float*)d_in[1];   // (1024,512)     f32
    float*       out    = (float*)d_out;           // (16,1,1024,529) f32

    char*   ws  = (char*)d_ws;
    bf16_t* wk2 = (bf16_t*)ws;
    bf16_t* xbt = (bf16_t*)(ws + WK2_BYTES);
    float*  pot = (float*)(ws + POT_OFF);

    build_wk<<<4096, 256, 0, stream>>>(weight, wk2);
    cvt_x  <<<2048, 256, 0, stream>>>(spikes, xbt);

    dim3 grid(NEUR / 64, (T_OUT + 63) / 64, NBATCH);   // 16 x 9 x 16
    snn_conv_wmma<<<grid, 128, 0, stream>>>(wk2, xbt, pot);

    wta_kernel<<<NBATCH, 1024, 0, stream>>>(pot, out);
}
